// FourierLayer2D_73349451481192
// MI455X (gfx1250) — matrix operand order repaired and checked by reading, not compiled, not run on hardware
//
// FNO spectral layer for MI455X (gfx1250): truncated-DFT-as-GEMM formulation.
// All matmul cores use v_wmma_f32_16x16x32_bf16 (f32 accumulate).
// B operands staged into LDS in fragment-major order (lane half h holds k 16h..16h+15) -> B frag = 2x ds_load_b128.
// Plain bf16 tile copies use GLOBAL_LOAD_ASYNC_TO_LDS_B128 (ASYNCcnt-tracked direct-to-LDS).
// Workspace requirement: ~84 MB (twiddles + complex intermediates, bf16).
#include <hip/hip_runtime.h>

#define B_  32
#define H_  128
#define W_  128
#define C_  64
#define MX_ 32
#define MY_ 32

typedef __attribute__((ext_vector_type(16))) __bf16 v16bf;
typedef __attribute__((ext_vector_type(8)))  float  v8f;
typedef __attribute__((ext_vector_type(4)))  __bf16 bf4;

// ---------------------------------------------------------------- async global->LDS (CDNA5)
#if defined(__gfx1250__) && __has_builtin(__builtin_amdgcn_global_load_async_to_lds_b128)
#define HAVE_ASYNC_LDS 1
#endif

#if HAVE_ASYNC_LDS
typedef int v4i_vs __attribute__((vector_size(16)));
typedef __attribute__((address_space(1))) v4i_vs* gptr16_t;
typedef __attribute__((address_space(3))) v4i_vs* lptr16_t;
static __device__ __forceinline__ void async_copy16(const void* g, void* l) {
  __builtin_amdgcn_global_load_async_to_lds_b128((gptr16_t)g, (lptr16_t)l, 0, 0);
}
static __device__ __forceinline__ void async_drain() {
#if __has_builtin(__builtin_amdgcn_s_wait_asynccnt)
  __builtin_amdgcn_s_wait_asynccnt(0);
#else
  asm volatile("s_wait_asynccnt 0x0" ::: "memory");
#endif
}
#endif

static __device__ __forceinline__ void stage_fence() {
#if HAVE_ASYNC_LDS
  async_drain();
#endif
}

static __device__ __forceinline__ v8f zero8() {
  v8f z;
#pragma unroll
  for (int i = 0; i < 8; ++i) z[i] = 0.0f;
  return z;
}

// ---------------------------------------------------------------- A fragment
// A (16x32, MxK) from row-major LDS tile, leading dim lda.
// lane<16: row=lane, K {0..15}; lane>=16: row=lane-16, K {16..31}.
// Rows are contiguous -> two 16B segments -> 2x ds_load_b128.
static __device__ __forceinline__ v16bf load_A_frag(const __bf16* Als, int lda) {
  const int lane = threadIdx.x & 31;
  const int m    = lane & 15;
  const int klo  = (lane & 16) ? 16 : 0;
  const __bf16* row = Als + m * lda;
  v16bf a;
#pragma unroll
  for (int j = 0; j < 8; ++j) a[j] = row[klo + j];
#pragma unroll
  for (int j = 0; j < 8; ++j) a[8 + j] = row[8 + klo + j];
  return a;
}

// ---------------------------------------------------------------- B fragment
// B stored fragment-major: per (kb,nb) 32x16 chunk of 512 bf16, laid out
// [lane][j]: lane = ((k&16)?16:0) | (n&15), j = k&15. Fragment load is one
// contiguous 32B per-lane read.
static __device__ __forceinline__ v16bf load_B_swz(const __bf16* base, int kb, int nb, int NB) {
  const int lane = threadIdx.x & 31;
  return *(const v16bf*)(base + (((kb * NB + nb) << 9) | (lane << 4)));
}

// Stage row-major bf16 global (K x N, row stride in elems) into swizzled LDS.
template <int K, int N>
static __device__ __forceinline__ void stage_B_swz(const __bf16* __restrict__ g, int grow_stride,
                                                   __bf16* __restrict__ l) {
  constexpr int VPR = N / 8;  // uint4 vectors per row
  const int total = K * VPR;
  for (int i = threadIdx.x; i < total; i += 256) {
    const int k = i / VPR, v = i - k * VPR;
    uint4 raw = *(const uint4*)(g + (size_t)k * grow_stride + v * 8);
    const __bf16* e = (const __bf16*)&raw;
    const int kb = k >> 5, kr = k & 31;
    const int j = kr & 15;
    const int halfsel = kr & 16;
    __bf16* chunkrow = l + ((kb * (N / 16)) << 9);
#pragma unroll
    for (int t = 0; t < 8; ++t) {
      const int n = v * 8 + t;
      chunkrow[(((n >> 4) << 9) | ((halfsel | (n & 15)) << 4) | j)] = e[t];
    }
  }
}

// Stage contiguous fp32 global (K x N) into swizzled bf16 LDS (cast on the fly).
template <int K, int N>
static __device__ __forceinline__ void stage_Bf32_swz(const float* __restrict__ g,
                                                      __bf16* __restrict__ l) {
  constexpr int VPR = N / 4;  // float4 vectors per row
  const int total = K * VPR;
  for (int i = threadIdx.x; i < total; i += 256) {
    const int k = i / VPR, v = i - k * VPR;
    float4 f = ((const float4*)g)[i];
    const int kb = k >> 5, kr = k & 31;
    const int j = kr & 15;
    const int halfsel = kr & 16;
    __bf16* chunkrow = l + ((kb * (N / 16)) << 9);
    const int n0 = v * 4;
    chunkrow[((((n0 + 0) >> 4) << 9) | ((halfsel | ((n0 + 0) & 15)) << 4) | j)] = (__bf16)f.x;
    chunkrow[((((n0 + 1) >> 4) << 9) | ((halfsel | ((n0 + 1) & 15)) << 4) | j)] = (__bf16)f.y;
    chunkrow[((((n0 + 2) >> 4) << 9) | ((halfsel | ((n0 + 2) & 15)) << 4) | j)] = (__bf16)f.z;
    chunkrow[((((n0 + 3) >> 4) << 9) | ((halfsel | ((n0 + 3) & 15)) << 4) | j)] = (__bf16)f.w;
  }
}

#define WMMA(a, b, c) \
  __builtin_amdgcn_wmma_f32_16x16x32_bf16(false, (a), false, (b), (short)0, (c), false, false)

// C/D layout: VGPR r -> row = r + 8*(lane>=16), col = lane&15.
static __device__ __forceinline__ void store_tile(__bf16* dst, int ld, v8f acc) {
  const int lane = threadIdx.x & 31;
  const int n  = lane & 15;
  const int mo = (lane & 16) ? 8 : 0;
#pragma unroll
  for (int r = 0; r < 8; ++r) dst[(size_t)(mo + r) * ld + n] = (__bf16)acc[r];
}

static __device__ __forceinline__ void store_tile2(__bf16* dst, int ld, v8f a, v8f b, float sb) {
  const int lane = threadIdx.x & 31;
  const int n  = lane & 15;
  const int mo = (lane & 16) ? 8 : 0;
#pragma unroll
  for (int r = 0; r < 8; ++r) dst[(size_t)(mo + r) * ld + n] = (__bf16)(a[r] + sb * b[r]);
}

// Cooperative contiguous 16B-vector copy (global bf16 -> LDS bf16, row-major A tiles).
static __device__ __forceinline__ void copy_vec(const void* __restrict__ g, void* __restrict__ l,
                                                int nvec) {
  const uint4* gs = (const uint4*)g;
  uint4* ls = (uint4*)l;
#if HAVE_ASYNC_LDS
  for (int i = threadIdx.x; i < nvec; i += 256) async_copy16(gs + i, ls + i);
#else
  for (int i = threadIdx.x; i < nvec; i += 256) ls[i] = gs[i];
#endif
}

// Strided-row copy for row-major A tiles.
template <int VPR>
static __device__ __forceinline__ void copy_rows(const void* __restrict__ g, void* __restrict__ l,
                                                 int rows, int grow_stride_vec) {
  const uint4* gs = (const uint4*)g;
  uint4* ls = (uint4*)l;
  const int total = rows * VPR;
  for (int i = threadIdx.x; i < total; i += 256) {
    int r = i / VPR, v = i - r * VPR;
#if HAVE_ASYNC_LDS
    async_copy16(gs + r * grow_stride_vec + v, ls + i);
#else
    ls[i] = gs[r * grow_stride_vec + v];
#endif
  }
}

// Stage fp32 global -> bf16 LDS row-major (A operands).
static __device__ __forceinline__ void stage_f32_bf16(const float* __restrict__ g,
                                                      __bf16* __restrict__ l, int nvec4) {
  const float4* gs = (const float4*)g;
  for (int i = threadIdx.x; i < nvec4; i += 256) {
    float4 v = gs[i];
    bf4 o;
    o.x = (__bf16)v.x; o.y = (__bf16)v.y; o.z = (__bf16)v.z; o.w = (__bf16)v.w;
    *(bf4*)(l + 4 * i) = o;
  }
}

// ---------------------------------------------------------------- twiddles
// Fc[k][t]=cos(2pi k t/128), Fs=-sin (forward e^{-i}); Ghr/Ghi = +cos/+sin (inverse);
// Gwr/Gwin = (cos, -sin)/(H*W)  (minus sign baked so stage 5 needs no negation).
__global__ __launch_bounds__(256) void k_twiddles(__bf16* Fc, __bf16* Fs, __bf16* Ghr, __bf16* Ghi,
                                                  __bf16* Gwr, __bf16* Gwin) {
  const float step = 6.28318530717958647692f / 128.0f;
  for (int i = threadIdx.x; i < MX_ * 128; i += blockDim.x) {
    {
      int k = i >> 7, t = i & 127;
      float th = step * (float)((k * t) & 127);
      Fc[i] = (__bf16)__cosf(th);
      Fs[i] = (__bf16)(-__sinf(th));
    }
    {
      int t = i >> 5, k = i & 31;
      float th = step * (float)((k * t) & 127);
      float c = __cosf(th), s = __sinf(th);
      Ghr[i] = (__bf16)c;
      Ghi[i] = (__bf16)s;
      const float inv = 1.0f / (float)(H_ * W_);
      Gwr[i]  = (__bf16)(c * inv);
      Gwin[i] = (__bf16)(-s * inv);
    }
  }
}

// ------------------------------------------------- stage 1: DFT along W
// per (b,h): X1[ky,c] = F(32x128) @ x[w,c](128x64), real input -> complex out.
__global__ __launch_bounds__(256) void k_dft_w(const float* __restrict__ x,
                                               const __bf16* __restrict__ Fc,
                                               const __bf16* __restrict__ Fs,
                                               __bf16* __restrict__ X1r, __bf16* __restrict__ X1i) {
  __shared__ __align__(16) __bf16 sFc[MX_ * W_];
  __shared__ __align__(16) __bf16 sFs[MX_ * W_];
  __shared__ __align__(32) __bf16 sX[W_ * C_];   // swizzled B
  const int b = blockIdx.x >> 7;
  const int h = blockIdx.x & 127;
  copy_vec(Fc, sFc, MX_ * W_ / 8);
  copy_vec(Fs, sFs, MX_ * W_ / 8);
  stage_Bf32_swz<W_, C_>(x + (size_t)(b * H_ + h) * (W_ * C_), sX);
  stage_fence();
  __syncthreads();
  const int wave = threadIdx.x >> 5;
  const int tr = wave >> 2;  // ky block 0..1
  const int tc = wave & 3;   // c  block 0..3
  v8f ar = zero8(), ai = zero8();
#pragma unroll
  for (int kk = 0; kk < W_; kk += 32) {
    v16bf bx = load_B_swz(sX, kk >> 5, tc, C_ / 16);
    ar = WMMA(load_A_frag(sFc + tr * 16 * W_ + kk, W_), bx, ar);
    ai = WMMA(load_A_frag(sFs + tr * 16 * W_ + kk, W_), bx, ai);
  }
  const size_t base = ((size_t)(b * H_ + h) * MY_ + tr * 16) * C_ + tc * 16;
  store_tile(X1r + base, C_, ar);
  store_tile(X1i + base, C_, ai);
}

// ------------------------------------------------- stage 2: DFT along H
// per (b,ky): X2[kx,c] = F(32x128) @ X1[h,c], complex x complex (4 real GEMMs).
__global__ __launch_bounds__(256) void k_dft_h(const __bf16* __restrict__ X1r,
                                               const __bf16* __restrict__ X1i,
                                               const __bf16* __restrict__ Fc,
                                               const __bf16* __restrict__ Fs,
                                               __bf16* __restrict__ X2r, __bf16* __restrict__ X2i) {
  __shared__ __align__(16) __bf16 sFc[MX_ * H_];
  __shared__ __align__(16) __bf16 sFs[MX_ * H_];
  __shared__ __align__(32) __bf16 sBr[H_ * C_];  // swizzled B
  __shared__ __align__(32) __bf16 sBi[H_ * C_];
  const int b  = blockIdx.x >> 5;
  const int ky = blockIdx.x & 31;
  copy_vec(Fc, sFc, MX_ * H_ / 8);
  copy_vec(Fs, sFs, MX_ * H_ / 8);
  const size_t sb = ((size_t)b * H_ * MY_ + ky) * C_;  // h=0 row
  stage_B_swz<H_, C_>(X1r + sb, MY_ * C_, sBr);
  stage_B_swz<H_, C_>(X1i + sb, MY_ * C_, sBi);
  stage_fence();
  __syncthreads();
  const int wave = threadIdx.x >> 5;
  const int tr = wave >> 2, tc = wave & 3;
  v8f rr = zero8(), ri = zero8(), ir = zero8(), ii = zero8();
#pragma unroll
  for (int kk = 0; kk < H_; kk += 32) {
    v16bf aC = load_A_frag(sFc + tr * 16 * H_ + kk, H_);
    v16bf aS = load_A_frag(sFs + tr * 16 * H_ + kk, H_);
    v16bf bR = load_B_swz(sBr, kk >> 5, tc, C_ / 16);
    v16bf bI = load_B_swz(sBi, kk >> 5, tc, C_ / 16);
    rr = WMMA(aC, bR, rr);
    ri = WMMA(aS, bI, ri);
    ir = WMMA(aS, bR, ir);
    ii = WMMA(aC, bI, ii);
  }
  // X2 index: ((b*MX+kx)*MY+ky)*C + c ; kx row stride = MY_*C_
  const size_t base = (size_t)b * (MX_ * MY_ * C_) + (size_t)(tr * 16) * (MY_ * C_) +
                      (size_t)ky * C_ + tc * 16;
  store_tile2(X2r + base, MY_ * C_, rr, ri, -1.0f);  // Re = Fc@r - Fs@i
  store_tile2(X2i + base, MY_ * C_, ii, ir, +1.0f);  // Im = Fc@i + Fs@r
}

// ------------------------------------------------- stage 3: per-mode channel mix
// per (kx,ky): X3[b,co] = X2[b,ci] @ Wm[ci,co], complex 32x64x64.
__global__ __launch_bounds__(256) void k_modes(const __bf16* __restrict__ X2r,
                                               const __bf16* __restrict__ X2i,
                                               const float* __restrict__ wre,
                                               const float* __restrict__ wim,
                                               __bf16* __restrict__ X3r, __bf16* __restrict__ X3i) {
  __shared__ __align__(16) __bf16 sAr[B_ * C_];
  __shared__ __align__(16) __bf16 sAi[B_ * C_];
  __shared__ __align__(32) __bf16 sBr[C_ * C_];  // swizzled weights
  __shared__ __align__(32) __bf16 sBi[C_ * C_];
  const int kx = blockIdx.x >> 5;
  const int ky = blockIdx.x & 31;
  const size_t abase = ((size_t)kx * MY_ + ky) * C_;  // b=0 row of X2
  copy_rows<8>(X2r + abase, sAr, B_, MX_ * MY_ * C_ / 8);
  copy_rows<8>(X2i + abase, sAi, B_, MX_ * MY_ * C_ / 8);
  const size_t wbase = ((size_t)kx * MY_ + ky) * (C_ * C_);
  stage_Bf32_swz<C_, C_>(wre + wbase, sBr);
  stage_Bf32_swz<C_, C_>(wim + wbase, sBi);
  stage_fence();
  __syncthreads();
  const int wave = threadIdx.x >> 5;
  const int tr = wave >> 2, tc = wave & 3;  // tr: b block, tc: co block
  v8f rr = zero8(), ri = zero8(), ir = zero8(), ii = zero8();
#pragma unroll
  for (int kk = 0; kk < C_; kk += 32) {
    v16bf aR = load_A_frag(sAr + tr * 16 * C_ + kk, C_);
    v16bf aI = load_A_frag(sAi + tr * 16 * C_ + kk, C_);
    v16bf bR = load_B_swz(sBr, kk >> 5, tc, C_ / 16);
    v16bf bI = load_B_swz(sBi, kk >> 5, tc, C_ / 16);
    rr = WMMA(aR, bR, rr);
    ri = WMMA(aI, bI, ri);
    ir = WMMA(aI, bR, ir);
    ii = WMMA(aR, bI, ii);
  }
  const size_t base = (size_t)(tr * 16) * (MX_ * MY_ * C_) + abase + tc * 16;
  store_tile2(X3r + base, MX_ * MY_ * C_, rr, ri, -1.0f);
  store_tile2(X3i + base, MX_ * MY_ * C_, ii, ir, +1.0f);
}

// ------------------------------------------------- stage 4: inverse DFT along H
// per (b,ky): T[h,c] = G(128x32) @ X3[kx,c], complex (K=32: single WMMA step).
__global__ __launch_bounds__(256) void k_idft_h(const __bf16* __restrict__ X3r,
                                                const __bf16* __restrict__ X3i,
                                                const __bf16* __restrict__ Ghr,
                                                const __bf16* __restrict__ Ghi,
                                                __bf16* __restrict__ Tr, __bf16* __restrict__ Ti) {
  __shared__ __align__(16) __bf16 sGr[H_ * MX_];
  __shared__ __align__(16) __bf16 sGi[H_ * MX_];
  __shared__ __align__(32) __bf16 sBr[MX_ * C_];  // swizzled B
  __shared__ __align__(32) __bf16 sBi[MX_ * C_];
  const int b  = blockIdx.x >> 5;
  const int ky = blockIdx.x & 31;
  copy_vec(Ghr, sGr, H_ * MX_ / 8);
  copy_vec(Ghi, sGi, H_ * MX_ / 8);
  const size_t bbase = (size_t)b * (MX_ * MY_ * C_) + (size_t)ky * C_;  // kx=0 row
  stage_B_swz<MX_, C_>(X3r + bbase, MY_ * C_, sBr);
  stage_B_swz<MX_, C_>(X3i + bbase, MY_ * C_, sBi);
  stage_fence();
  __syncthreads();
  const int wave = threadIdx.x >> 5;
#pragma unroll
  for (int it = 0; it < 4; ++it) {
    const int tile = it * 8 + wave;
    const int tr = tile >> 2, tc = tile & 3;  // tr: h block 0..7, tc: c block
    v16bf aR = load_A_frag(sGr + tr * 16 * MX_, MX_);
    v16bf aI = load_A_frag(sGi + tr * 16 * MX_, MX_);
    v16bf bR = load_B_swz(sBr, 0, tc, C_ / 16);
    v16bf bI = load_B_swz(sBi, 0, tc, C_ / 16);
    v8f rr = WMMA(aR, bR, zero8());
    v8f ri = WMMA(aI, bI, zero8());
    v8f ir = WMMA(aI, bR, zero8());
    v8f ii = WMMA(aR, bI, zero8());
    const size_t base = (size_t)b * (H_ * MY_ * C_) + (size_t)(tr * 16) * (MY_ * C_) +
                        (size_t)ky * C_ + tc * 16;
    store_tile2(Tr + base, MY_ * C_, rr, ri, -1.0f);
    store_tile2(Ti + base, MY_ * C_, ii, ir, +1.0f);
  }
}

// ---------------------- stage 5: inverse DFT along W + local conv + bias + ReLU
// per (b,h): y[w,co] = relu( Gwr@Tr + Gwin@Ti + x[w,ci]@convW + bias ), fp32 out.
__global__ __launch_bounds__(256) void k_final(const float* __restrict__ x,
                                               const float* __restrict__ convw,
                                               const float* __restrict__ convb,
                                               const __bf16* __restrict__ Gwr,
                                               const __bf16* __restrict__ Gwin,
                                               const __bf16* __restrict__ Tr,
                                               const __bf16* __restrict__ Ti,
                                               float* __restrict__ out) {
  __shared__ __align__(16) __bf16 sGr[W_ * MY_];
  __shared__ __align__(16) __bf16 sGi[W_ * MY_];
  __shared__ __align__(32) __bf16 sTr[MY_ * C_];  // swizzled B
  __shared__ __align__(32) __bf16 sTi[MY_ * C_];
  __shared__ __align__(16) __bf16 sX[W_ * C_];    // row-major A
  __shared__ __align__(32) __bf16 sW[C_ * C_];    // swizzled B
  __shared__ float sBias[C_];
  const int b = blockIdx.x >> 7;
  const int h = blockIdx.x & 127;
  copy_vec(Gwr, sGr, W_ * MY_ / 8);
  copy_vec(Gwin, sGi, W_ * MY_ / 8);
  const size_t tbase = (size_t)(b * H_ + h) * (MY_ * C_);  // contiguous (ky,c) slab
  stage_B_swz<MY_, C_>(Tr + tbase, C_, sTr);
  stage_B_swz<MY_, C_>(Ti + tbase, C_, sTi);
  stage_f32_bf16(x + (size_t)(b * H_ + h) * (W_ * C_), sX, W_ * C_ / 4);
  stage_Bf32_swz<C_, C_>(convw, sW);
  if (threadIdx.x < C_) sBias[threadIdx.x] = convb[threadIdx.x];
  stage_fence();
  __syncthreads();
  const int wave = threadIdx.x >> 5;
  const int lane = threadIdx.x & 31;
  const int n  = lane & 15;
  const int mo = (lane & 16) ? 8 : 0;
#pragma unroll
  for (int it = 0; it < 4; ++it) {
    const int tile = it * 8 + wave;
    const int tr = tile >> 2, tc = tile & 3;  // tr: w block 0..7, tc: co block
    v8f acc = zero8();
    acc = WMMA(load_A_frag(sGr + tr * 16 * MY_, MY_), load_B_swz(sTr, 0, tc, C_ / 16), acc);
    acc = WMMA(load_A_frag(sGi + tr * 16 * MY_, MY_), load_B_swz(sTi, 0, tc, C_ / 16), acc);
#pragma unroll
    for (int kk = 0; kk < C_; kk += 32) {
      acc = WMMA(load_A_frag(sX + tr * 16 * C_ + kk, C_),
                 load_B_swz(sW, kk >> 5, tc, C_ / 16), acc);
    }
    const float bias = sBias[tc * 16 + n];
    float* o = out + (size_t)(b * H_ + h) * (W_ * C_) + (size_t)(tr * 16) * C_ + tc * 16;
#pragma unroll
    for (int r = 0; r < 8; ++r) {
      float v = acc[r] + bias;
      o[(size_t)(mo + r) * C_ + n] = v > 0.0f ? v : 0.0f;
    }
  }
}

extern "C" void kernel_launch(void* const* d_in, const int* in_sizes, int n_in,
                              void* d_out, int out_size, void* d_ws, size_t ws_size,
                              hipStream_t stream) {
  (void)in_sizes; (void)n_in; (void)out_size; (void)ws_size;
  const float* x  = (const float*)d_in[0];
  const float* wr = (const float*)d_in[1];
  const float* wi = (const float*)d_in[2];
  const float* cw = (const float*)d_in[3];
  const float* cb = (const float*)d_in[4];
  float* out = (float*)d_out;

  char* ws = (char*)d_ws;
  size_t off = 0;
  auto take = [&](size_t bytes) -> void* {
    void* q = ws + off;
    off += (bytes + 255) & ~(size_t)255;
    return q;
  };
  __bf16* Fc   = (__bf16*)take((size_t)MX_ * W_ * 2);
  __bf16* Fs   = (__bf16*)take((size_t)MX_ * W_ * 2);
  __bf16* Ghr  = (__bf16*)take((size_t)H_ * MX_ * 2);
  __bf16* Ghi  = (__bf16*)take((size_t)H_ * MX_ * 2);
  __bf16* Gwr  = (__bf16*)take((size_t)W_ * MY_ * 2);
  __bf16* Gwin = (__bf16*)take((size_t)W_ * MY_ * 2);
  __bf16* X1r  = (__bf16*)take((size_t)B_ * H_ * MY_ * C_ * 2);
  __bf16* X1i  = (__bf16*)take((size_t)B_ * H_ * MY_ * C_ * 2);
  __bf16* X2r  = (__bf16*)take((size_t)B_ * MX_ * MY_ * C_ * 2);
  __bf16* X2i  = (__bf16*)take((size_t)B_ * MX_ * MY_ * C_ * 2);
  __bf16* X3r  = (__bf16*)take((size_t)B_ * MX_ * MY_ * C_ * 2);
  __bf16* X3i  = (__bf16*)take((size_t)B_ * MX_ * MY_ * C_ * 2);
  __bf16* Trm  = (__bf16*)take((size_t)B_ * H_ * MY_ * C_ * 2);
  __bf16* Tim  = (__bf16*)take((size_t)B_ * H_ * MY_ * C_ * 2);

  k_twiddles<<<1, 256, 0, stream>>>(Fc, Fs, Ghr, Ghi, Gwr, Gwin);
  k_dft_w <<<B_ * H_,   256, 0, stream>>>(x, Fc, Fs, X1r, X1i);
  k_dft_h <<<B_ * MY_,  256, 0, stream>>>(X1r, X1i, Fc, Fs, X2r, X2i);
  k_modes <<<MX_ * MY_, 256, 0, stream>>>(X2r, X2i, wr, wi, X3r, X3i);
  k_idft_h<<<B_ * MY_,  256, 0, stream>>>(X3r, X3i, Ghr, Ghi, Trm, Tim);
  k_final <<<B_ * H_,   256, 0, stream>>>(x, cw, cb, Gwr, Gwin, Trm, Tim, out);
}
